// TensorMemoryLayer_30073361007090
// MI455X (gfx1250) — matrix operand order repaired and checked by reading, not compiled, not run on hardware
//
#include <hip/hip_runtime.h>
#include <hip/hip_bf16.h>
#include <hip/hip_fp16.h>

typedef _Float16 v16h __attribute__((ext_vector_type(16)));
typedef float    v8f  __attribute__((ext_vector_type(8)));

#define B_   4
#define T_   2048
#define D_   1024
#define F_   4096
#define H_   4
#define HD_  256
#define CH_  16      // attention chunk length
#define ES_  64      // e-dim slab of M held per workgroup
#define EPSM 1e-6f

// ---- A-fragment K permutation (involution): storage pos <-> natural K within a 32-block
//   pos [0..7]=K0..7, [8..15]=K16..23, [16..23]=K8..15, [24..31]=K24..31
// With this storage order, lane-half kh reads its whole v16h fragment from
// 16 contiguous halves at offset 16*kh  ->  2x ds_load_b128.
__device__ __forceinline__ int permk(int p) {           // pos->natural K (and K->pos)
    return (p < 8) ? p : (p < 16 ? p + 8 : (p < 24 ? p - 8 : p));
}
__device__ __forceinline__ int permc8(int c8) {         // chunk-base map for 8-wide stores
    return (c8 == 8) ? 16 : (c8 == 16 ? 8 : c8);
}

// 32B fragment load from 16B-aligned LDS (2 x b128)
__device__ __forceinline__ v16h ld_frag32(const _Float16* p) {
    union { uint4 u[2]; v16h h; } t;
    t.u[0] = *reinterpret_cast<const uint4*>(p);
    t.u[1] = *reinterpret_cast<const uint4*>(p + 8);
    return t.h;
}

__device__ __forceinline__ float gelu_tanh(float x) {
    float x3 = x * x * x;
    return 0.5f * x * (1.0f + tanhf(0.7978845608028654f * (x + 0.044715f * x3)));
}

// ============================ fp32 -> f16 cast ============================
__global__ void cast_f32_f16(const float* __restrict__ in, _Float16* __restrict__ out, int n) {
    for (int i = blockIdx.x * blockDim.x + threadIdx.x; i < n; i += gridDim.x * blockDim.x)
        out[i] = (_Float16)in[i];
}

// ============================ LayerNorm -> f16 ============================
__global__ __launch_bounds__(256) void ln_cast(const float* __restrict__ x,
                                               const float* __restrict__ g,
                                               const float* __restrict__ bb,
                                               _Float16* __restrict__ out) {
    __shared__ float rs[256], rs2[256];
    const int row = blockIdx.x;
    const float* xr = x + (size_t)row * D_;
    float s = 0.f, s2 = 0.f;
    for (int i = threadIdx.x; i < D_; i += 256) { float v = xr[i]; s += v; s2 += v * v; }
    rs[threadIdx.x] = s; rs2[threadIdx.x] = s2;
    __syncthreads();
    for (int st = 128; st > 0; st >>= 1) {
        if (threadIdx.x < st) { rs[threadIdx.x] += rs[threadIdx.x + st]; rs2[threadIdx.x] += rs2[threadIdx.x + st]; }
        __syncthreads();
    }
    const float mu  = rs[0] * (1.0f / D_);
    const float var = rs2[0] * (1.0f / D_) - mu * mu;
    const float inv = rsqrtf(var + 1e-5f);
    for (int i = threadIdx.x; i < D_; i += 256)
        out[(size_t)row * D_ + i] = (_Float16)((xr[i] - mu) * inv * g[i] + bb[i]);
}

// ============================ WMMA GEMM =================================
// C[M,N] = act(A[M,K](f16) @ W[K,N](f16) + bias) (+ res) -> f32 or f16
// 256 thr (8 waves), tile 128x128, K step 32; wave grid 4(M)x2(N).
// As: natural-K rows  -> A fragment = 2 x ds_load_b128
// Bt: transposed tile  -> B fragment = 2 x ds_load_b128
template <int ACT, bool ADD_RES, bool OUT_F16>
__global__ __launch_bounds__(256) void gemm_wmma(const _Float16* __restrict__ A,
                                                 const _Float16* __restrict__ W,
                                                 const float* __restrict__ bias,
                                                 const float* __restrict__ res,
                                                 float* __restrict__ outF,
                                                 _Float16* __restrict__ outH,
                                                 int M, int N, int K) {
    __shared__ __align__(16) _Float16 As[128 * 40];   // 128 rows x 32 K (natural) + pad
    __shared__ __align__(16) _Float16 Bt[128 * 40];   // 128 N-rows x 32 K + pad

    const int tid  = threadIdx.x;
    const int lane = tid & 31, wid = tid >> 5;
    const int waveM = wid & 3, waveN = wid >> 2;       // 4 x 2
    const int bm = blockIdx.y * 128, bn = blockIdx.x * 128;
    const int nn = lane & 15, kh = lane >> 4;

    v8f acc[2][4];
#pragma unroll
    for (int i = 0; i < 2; ++i)
#pragma unroll
        for (int j = 0; j < 4; ++j)
#pragma unroll
            for (int r = 0; r < 8; ++r) acc[i][j][r] = 0.0f;

    for (int k0 = 0; k0 < K; k0 += 32) {
#pragma unroll
        for (int it = 0; it < 2; ++it) {
            int idx = it * 256 + tid;                  // 0..511
            // A: 128 x 32, store 8-chunk at natural K base
            int r = idx >> 2, c8 = (idx & 3) * 8;
            *reinterpret_cast<uint4*>(&As[r * 40 + c8]) =
                *reinterpret_cast<const uint4*>(&A[(size_t)(bm + r) * K + k0 + c8]);
            // B: 32 x 128 in memory -> transpose into Bt[n][k]
            int rb = idx >> 4, cb8 = (idx & 15) * 8;
            union { uint4 u; _Float16 h[8]; } t;
            t.u = *reinterpret_cast<const uint4*>(&W[(size_t)(k0 + rb) * N + bn + cb8]);
#pragma unroll
            for (int j = 0; j < 8; ++j) Bt[(cb8 + j) * 40 + rb] = t.h[j];
        }
        if (k0 + 32 < K) {   // prefetch next K tiles -> global_prefetch_b8
            __builtin_prefetch(&A[(size_t)(bm + (tid >> 1)) * K + k0 + 32], 0, 0);
            __builtin_prefetch(&W[(size_t)(k0 + 32 + (tid >> 3)) * N + bn + (tid & 7) * 16], 0, 0);
        }
        __syncthreads();

        v16h afrag[2], bfrag[4];
#pragma unroll
        for (int i = 0; i < 2; ++i)
            afrag[i] = ld_frag32(&As[(waveM * 32 + i * 16 + nn) * 40 + kh * 16]);
#pragma unroll
        for (int j = 0; j < 4; ++j)
            bfrag[j] = ld_frag32(&Bt[(waveN * 64 + j * 16 + nn) * 40 + kh * 16]);
#pragma unroll
        for (int i = 0; i < 2; ++i)
#pragma unroll
            for (int j = 0; j < 4; ++j)
                acc[i][j] = __builtin_amdgcn_wmma_f32_16x16x32_f16(
                    false, afrag[i], false, bfrag[j], (short)0, acc[i][j], false, false);
        __syncthreads();
    }

    // ---- epilogue: C layout: VGPR r -> row r+8*kh, col = lane&15
#pragma unroll
    for (int i = 0; i < 2; ++i)
#pragma unroll
        for (int j = 0; j < 4; ++j) {
            const int mb = bm + waveM * 32 + i * 16;
            const int nb = bn + waveN * 64 + j * 16;
#pragma unroll
            for (int r = 0; r < 8; ++r) {
                const int gr = mb + r + 8 * kh;
                const int gc = nb + nn;
                float v = acc[i][j][r] + bias[gc];
                if (ACT == 1) v = gelu_tanh(v);
                if (ADD_RES) v += res[(size_t)gr * N + gc];
                if (OUT_F16) outH[(size_t)gr * N + gc] = (_Float16)v;
                else         outF[(size_t)gr * N + gc] = v;
            }
        }
}

// ====================== chunked linear-attention scan ======================
// grid = B*H*(HD/ES) = 64 workgroups; block = 256 (8 waves).
// M slab [HD x ES] f32 (accumulator master) + transposed f16 mirror M16t[ES x HD]
// both live in LDS -- CDNA5 320KB/WGP.
#define ATT_SMEM ((HD_ * ES_ + HD_ + 256 + 16) * 4 + \
                  (2 * CH_ * HD_ + CH_ * ES_ + CH_ * 32 + ES_ * HD_) * 2)

__global__ __launch_bounds__(256) void attn_kernel(const _Float16* __restrict__ qg,
                                                   const _Float16* __restrict__ kg,
                                                   const _Float16* __restrict__ vg,
                                                   _Float16* __restrict__ og) {
    extern __shared__ char smem_raw[];
    float*     Msh  = (float*)smem_raw;            // HD x ES (f32 master)
    float*     zsh  = Msh + HD_ * ES_;             // HD
    float*     Sf   = zsh + HD_;                   // 16x16
    float*     den  = Sf + 256;                    // 16
    _Float16*  Qc   = (_Float16*)(den + 16);       // 16 x HD (natural)
    _Float16*  Kc   = Qc + CH_ * HD_;              // 16 x HD (natural)
    _Float16*  Vc   = Kc + CH_ * HD_;              // 16 x ES
    _Float16*  Sm   = Vc + CH_ * ES_;              // 16 x 32 (natural, zero-padded)
    _Float16*  M16t = Sm + CH_ * 32;               // ES x HD (f16 mirror of M^T)

    const int tid  = threadIdx.x;
    const int lane = tid & 31, wid = tid >> 5;
    const int nn = lane & 15, kh = lane >> 4;

    const int wg   = blockIdx.x;
    const int slab = wg & 3;
    const int h    = (wg >> 2) & 3;
    const int b    = wg >> 4;
    const int e0   = slab * ES_;

    for (int i = tid; i < HD_ * ES_; i += 256) Msh[i] = 0.0f;
    for (int i = tid; i < ES_ * HD_ / 8; i += 256)
        *reinterpret_cast<uint4*>(&M16t[i * 8]) = uint4{0, 0, 0, 0};
    if (tid < HD_) zsh[tid] = 0.0f;
    __syncthreads();

    const size_t baseRow = ((size_t)b * T_) * D_ + (size_t)h * HD_;

    for (int t0 = 0; t0 < T_; t0 += CH_) {
        // ---- P1: load Q (natural), K, V; den := eps
#pragma unroll
        for (int it = 0; it < 2; ++it) {
            int idx = it * 256 + tid;                    // 0..511
            int t = idx >> 5, c8 = (idx & 31) * 8;
            size_t g = baseRow + (size_t)(t0 + t) * D_ + c8;
            int qpos = c8;
            *reinterpret_cast<uint4*>(&Qc[t * HD_ + qpos]) = *reinterpret_cast<const uint4*>(&qg[g]);
            *reinterpret_cast<uint4*>(&Kc[t * HD_ + c8])   = *reinterpret_cast<const uint4*>(&kg[g]);
        }
        {
            int t = tid >> 4, c4 = (tid & 15) * 4;
            size_t g = baseRow + (size_t)(t0 + t) * D_ + e0 + c4;
            *reinterpret_cast<unsigned long long*>(&Vc[t * ES_ + c4]) =
                *reinterpret_cast<const unsigned long long*>(&vg[g]);
        }
        if (tid < CH_) den[tid] = EPSM;
        __syncthreads();

        // ---- P2: S = Q K^T  (wave 0, 8 WMMAs over K=HD)
        if (wid == 0) {
            v8f s;
#pragma unroll
            for (int r = 0; r < 8; ++r) s[r] = 0.0f;
#pragma unroll
            for (int k = 0; k < HD_ / 32; ++k) {
                const int kb = k * 32;
                v16h a  = ld_frag32(&Qc[nn * HD_ + kb + 16 * kh]);   // natural rows
                v16h bf = ld_frag32(&Kc[nn * HD_ + kb + 16 * kh]);   // B[k][n]=K[n][k], natural
                s = __builtin_amdgcn_wmma_f32_16x16x32_f16(false, a, false, bf, (short)0, s, false, false);
            }
#pragma unroll
            for (int r = 0; r < 8; ++r) Sf[(r + 8 * kh) * 16 + nn] = s[r];
        }
        __syncthreads();

        // ---- P3: masked S -> f16 at natural pos; den += q.z_prev + rowsum
#pragma unroll
        for (int it = 0; it < 2; ++it) {
            int idx = it * 256 + tid;
            int r = idx >> 5, c = idx & 31;              // natural K position c
            float vl = (c < 16 && c <= r) ? Sf[r * 16 + c] : 0.0f;
            Sm[r * 32 + c] = (_Float16)vl;
        }
        {
            int r = tid >> 4, p = tid & 15;
            float pd = 0.0f;
#pragma unroll
            for (int d = 0; d < 16; ++d) {
                int q = p * 16 + d;                      // natural storage position
                int zi = q;                              // natural K index
                pd += (float)Qc[r * HD_ + q] * zsh[zi];
            }
            if (p == 0)
                for (int s2 = 0; s2 <= r; ++s2) pd += Sf[r * 16 + s2];
            atomicAdd(&den[r], pd);
        }
        __syncthreads();

        // ---- P4: out = Q@M_prev + mask(S)@V, /den  (waves 0..3)
        if (wid < 4) {
            const int nb = wid * 16;
            v8f acc;
#pragma unroll
            for (int r = 0; r < 8; ++r) acc[r] = 0.0f;
#pragma unroll
            for (int k = 0; k < HD_ / 32; ++k) {
                const int kb = k * 32;
                v16h a  = ld_frag32(&Qc[nn * HD_ + kb + 16 * kh]);
                v16h bf = ld_frag32(&M16t[(nb + nn) * HD_ + kb + 16 * kh]);  // M^T mirror
                acc = __builtin_amdgcn_wmma_f32_16x16x32_f16(false, a, false, bf, (short)0, acc, false, false);
            }
            {
                v16h a = ld_frag32(&Sm[nn * 32 + 16 * kh]);                  // natural S
                v16h bf;
#pragma unroll
                for (int v = 0; v < 8; ++v) {
                    const int k2 = 16 * kh + 2 * v;
                    _Float16 b0 = (_Float16)0.0f, b1 = (_Float16)0.0f;
                    if (k2 < 16) { b0 = Vc[k2 * ES_ + nb + nn]; b1 = Vc[(k2 + 1) * ES_ + nb + nn]; }
                    bf[2 * v] = b0; bf[2 * v + 1] = b1;
                }
                acc = __builtin_amdgcn_wmma_f32_16x16x32_f16(false, a, false, bf, (short)0, acc, false, false);
            }
#pragma unroll
            for (int r = 0; r < 8; ++r) {
                const int trow = r + 8 * kh;
                float vl = acc[r] / den[trow];
                og[baseRow + (size_t)(t0 + trow) * D_ + e0 + nb + nn] = (_Float16)vl;
            }
        }
        __syncthreads();

        // ---- P5: M += K^T V (64 tiles, 8/wave); update f16 mirror; z += sum_t k_t
#pragma unroll
        for (int i = 0; i < 8; ++i) {
            const int tI = wid * 8 + i;                 // 0..63
            const int d0 = (tI >> 2) * 16, nb2 = (tI & 3) * 16;
            v8f c;
#pragma unroll
            for (int r = 0; r < 8; ++r) c[r] = Msh[(d0 + r + 8 * kh) * ES_ + nb2 + nn];
            v16h a, bf;
#pragma unroll
            for (int v = 0; v < 8; ++v) {
                // A = K^T: row=d, K-dim=t (valid t<16, zero-pad)
                const int kkA = 16 * kh + 2 * v;
                _Float16 a0 = (_Float16)0.0f, a1 = (_Float16)0.0f;
                if (kkA < 16)     a0 = Kc[kkA * HD_ + d0 + nn];
                if (kkA + 1 < 16) a1 = Kc[(kkA + 1) * HD_ + d0 + nn];
                a[2 * v] = a0; a[2 * v + 1] = a1;
                const int k2 = 16 * kh + 2 * v;
                _Float16 b0 = (_Float16)0.0f, b1 = (_Float16)0.0f;
                if (k2 < 16) { b0 = Vc[k2 * ES_ + nb2 + nn]; b1 = Vc[(k2 + 1) * ES_ + nb2 + nn]; }
                bf[2 * v] = b0; bf[2 * v + 1] = b1;
            }
            c = __builtin_amdgcn_wmma_f32_16x16x32_f16(false, a, false, bf, (short)0, c, false, false);
#pragma unroll
            for (int r = 0; r < 8; ++r) Msh[(d0 + r + 8 * kh) * ES_ + nb2 + nn] = c[r];
            union { uint4 u; _Float16 hh[8]; } pk;
#pragma unroll
            for (int r = 0; r < 8; ++r) pk.hh[r] = (_Float16)c[r];
            *reinterpret_cast<uint4*>(&M16t[(nb2 + nn) * HD_ + d0 + 8 * kh]) = pk.u;  // M^T mirror
        }
        {
            float s = 0.0f;
#pragma unroll
            for (int t = 0; t < CH_; ++t) s += (float)Kc[t * HD_ + tid];
            zsh[tid] += s;
        }
        __syncthreads();
    }
}

// ================================ launch ================================
extern "C" void kernel_launch(void* const* d_in, const int* in_sizes, int n_in,
                              void* d_out, int out_size, void* d_ws, size_t ws_size,
                              hipStream_t stream) {
    const float* x    = (const float*)d_in[0];
    const float* ln1g = (const float*)d_in[1];
    const float* ln1b = (const float*)d_in[2];
    const float* ln2g = (const float*)d_in[3];
    const float* ln2b = (const float*)d_in[4];
    const float* Wq = (const float*)d_in[5];  const float* bq = (const float*)d_in[6];
    const float* Wk = (const float*)d_in[7];  const float* bk = (const float*)d_in[8];
    const float* Wv = (const float*)d_in[9];  const float* bv = (const float*)d_in[10];
    const float* Wo = (const float*)d_in[11]; const float* bo = (const float*)d_in[12];
    const float* W1 = (const float*)d_in[13]; const float* b1 = (const float*)d_in[14];
    const float* W2 = (const float*)d_in[15]; const float* b2 = (const float*)d_in[16];

    char* ws = (char*)d_ws;
    size_t off = 0;
    auto take = [&](size_t bytes) -> char* {
        char* p = ws + off;
        off = (off + bytes + 255) & ~(size_t)255;
        return p;
    };
    const size_t MT = (size_t)B_ * T_;
    _Float16* Wq16 = (_Float16*)take((size_t)D_ * D_ * 2);
    _Float16* Wk16 = (_Float16*)take((size_t)D_ * D_ * 2);
    _Float16* Wv16 = (_Float16*)take((size_t)D_ * D_ * 2);
    _Float16* Wo16 = (_Float16*)take((size_t)D_ * D_ * 2);
    _Float16* W116 = (_Float16*)take((size_t)D_ * F_ * 2);
    _Float16* W216 = (_Float16*)take((size_t)F_ * D_ * 2);
    _Float16* h16  = (_Float16*)take(MT * D_ * 2);     // reused for LN2 output
    _Float16* q16  = (_Float16*)take(MT * D_ * 2);
    _Float16* k16  = (_Float16*)take(MT * D_ * 2);
    _Float16* v16  = (_Float16*)take(MT * D_ * 2);
    _Float16* at16 = (_Float16*)take(MT * D_ * 2);
    float*    x1   = (float*)take(MT * D_ * 4);
    _Float16* mid16= (_Float16*)take(MT * F_ * 2);
    (void)ws_size; (void)n_in; (void)in_sizes; (void)out_size;

    (void)hipFuncSetAttribute(reinterpret_cast<const void*>(attn_kernel),
                              hipFuncAttributeMaxDynamicSharedMemorySize, ATT_SMEM);

    // weights -> f16
    cast_f32_f16<<<1024, 256, 0, stream>>>(Wq, Wq16, D_ * D_);
    cast_f32_f16<<<1024, 256, 0, stream>>>(Wk, Wk16, D_ * D_);
    cast_f32_f16<<<1024, 256, 0, stream>>>(Wv, Wv16, D_ * D_);
    cast_f32_f16<<<1024, 256, 0, stream>>>(Wo, Wo16, D_ * D_);
    cast_f32_f16<<<2048, 256, 0, stream>>>(W1, W116, D_ * F_);
    cast_f32_f16<<<2048, 256, 0, stream>>>(W2, W216, F_ * D_);

    // h = LN1(x) (f16)
    ln_cast<<<B_ * T_, 256, 0, stream>>>(x, ln1g, ln1b, h16);

    // q,k,v = h @ W + b (f16)
    dim3 gD(D_ / 128, (B_ * T_) / 128);
    gemm_wmma<0, false, true><<<gD, 256, 0, stream>>>(h16, Wq16, bq, nullptr, nullptr, q16, B_ * T_, D_, D_);
    gemm_wmma<0, false, true><<<gD, 256, 0, stream>>>(h16, Wk16, bk, nullptr, nullptr, k16, B_ * T_, D_, D_);
    gemm_wmma<0, false, true><<<gD, 256, 0, stream>>>(h16, Wv16, bv, nullptr, nullptr, v16, B_ * T_, D_, D_);

    // chunked causal linear-attention scan (M slab in LDS)
    attn_kernel<<<B_ * H_ * (HD_ / ES_), 256, ATT_SMEM, stream>>>(q16, k16, v16, at16);

    // x1 = x + attn @ Wo + bo (f32)
    gemm_wmma<0, true, false><<<gD, 256, 0, stream>>>(at16, Wo16, bo, x, x1, nullptr, B_ * T_, D_, D_);

    // h2 = LN2(x1) (f16, reuse h16)
    ln_cast<<<B_ * T_, 256, 0, stream>>>(x1, ln2g, ln2b, h16);

    // mid = gelu(h2 @ W1 + b1) (f16)
    dim3 gF(F_ / 128, (B_ * T_) / 128);
    gemm_wmma<1, false, true><<<gF, 256, 0, stream>>>(h16, W116, b1, nullptr, nullptr, mid16, B_ * T_, F_, D_);

    // out = x1 + mid @ W2 + b2 (f32)
    gemm_wmma<0, true, false><<<gD, 256, 0, stream>>>(mid16, W216, b2, x1, (float*)d_out, nullptr, B_ * T_, D_, F_);
}
